// Assembler_31447750542013
// MI455X (gfx1250) — matrix operand order repaired and checked by reading, not compiled, not run on hardware
//
#include <hip/hip_runtime.h>

// ============================================================================
// MI455X (gfx1250) forward for the response-assembler transformer.
// ~700 GFLOP, compute-bound (23.3 TB/s HBM needs only ~26us for the ~600MB of
// weights); all large GEMMs run on v_wmma_f32_16x16x32_bf16 (bf16 operands,
// fp32 accumulate), converting fp32 operands to bf16 during the global->LDS
// tile stage. Each wave owns a 32x32 output tile: 4 WMMAs per 8 ds_load_b128
// (2.0 LDS loads per WMMA), block tile 64x128 with 8 wave32s.
// Workspace: ~294 MB fp32.
// ============================================================================

typedef __bf16 bf16;
typedef bf16  v16bf __attribute__((ext_vector_type(16)));
typedef bf16  v8bf  __attribute__((ext_vector_type(8)));
typedef float v8f   __attribute__((ext_vector_type(8)));

#define TPB 256
#define LSTR 40   // LDS row stride in bf16 (80B): conflict-free 16B frag reads

namespace {

constexpr int Dm  = 1024;   // model dim
constexpr int Hh  = 8;      // heads
constexpr int HDd = 128;    // head dim
constexpr int BBt = 2;      // batch
constexpr int SQs = 1024;   // query seq
constexpr int SKVs= 512;    // per-response kv seq
constexpr int NRs = 4;      // responses
constexpr int KVT = 2048;   // NRs*SKVs
constexpr int NLl = 6;      // generator layers
constexpr int FFd = 4096;
constexpr int VOC = 32000;
constexpr float EPSf = 1e-6f;
constexpr float ATT_SCALE = 0.08838834764831845f;  // 1/sqrt(128)

// ---------------- block reductions ----------------
__device__ __forceinline__ float bsum(float v, float* sd) {
  int t = threadIdx.x; sd[t] = v; __syncthreads();
  #pragma unroll
  for (int s = TPB/2; s > 0; s >>= 1) { if (t < s) sd[t] += sd[t+s]; __syncthreads(); }
  float r = sd[0]; __syncthreads(); return r;
}
__device__ __forceinline__ float bmaxr(float v, float* sd) {
  int t = threadIdx.x; sd[t] = v; __syncthreads();
  #pragma unroll
  for (int s = TPB/2; s > 0; s >>= 1) { if (t < s) sd[t] = fmaxf(sd[t], sd[t+s]); __syncthreads(); }
  float r = sd[0]; __syncthreads(); return r;
}

// ---------------- WMMA fragment loader ----------------
__device__ __forceinline__ v16bf fragA(const bf16* rowbase, int hlf) {
  // A frag: lane holds row m=lane&15; lanes0-15 K=0..15, lanes16-31 K=16..31
  v8bf lo = *(const v8bf*)(rowbase + hlf * 16);
  v8bf hi = *(const v8bf*)(rowbase + hlf * 16 + 8);
  return __builtin_shufflevector(lo, hi, 0,1,2,3,4,5,6,7,8,9,10,11,12,13,14,15);
}
__device__ __forceinline__ v16bf fragB(const bf16* rowbase, int hlf) {
  // B frag: lane holds column n=lane&15; lanes0-15 K=0..15, lanes16-31 K=16..31
  v8bf lo = *(const v8bf*)(rowbase + hlf * 16);
  v8bf hi = *(const v8bf*)(rowbase + hlf * 16 + 8);
  return __builtin_shufflevector(lo, hi, 0,1,2,3,4,5,6,7,8,9,10,11,12,13,14,15);
}

// ---------------------------------------------------------------------------
// WMMA GEMM:  C[m,n] = alpha * sum_k A[m,k]*Bop[k,n] (+ bias[n])
//  TRANSB=true : B is [N,K] row-major (x@W.T, q@k.T);  false: B is [K,N] (p@v)
//  Two-level batch: z = blockIdx.z; offset (z/Hb)*s?b + (z%Hb)*s?h.
//  256 thr = 8 wave32s (2x4), block tile M64 x N128, K chunk 32.
//  Each wave: 32x32 outputs = 4 accumulators, 4 WMMAs / k-chunk.
// ---------------------------------------------------------------------------
template<bool TRANSB>
__global__ __launch_bounds__(TPB)
void kgemm(const float* __restrict__ A, long long lda, long long sAb, long long sAh,
           const float* __restrict__ Bm, long long ldb, long long sBb, long long sBh,
           float* __restrict__ C, long long ldc, long long sCb, long long sCh,
           int K, const float* __restrict__ bias, float alpha, int Hb)
{
  __shared__ __align__(32) bf16 As[ 64 * LSTR];   //  5,120 B
  __shared__ __align__(32) bf16 Bs[128 * LSTR];   // 10,240 B

  const int z = blockIdx.z, zb = z / Hb, zh = z % Hb;
  A  += (long long)zb * sAb + (long long)zh * sAh;
  Bm += (long long)zb * sBb + (long long)zh * sBh;
  C  += (long long)zb * sCb + (long long)zh * sCh;

  const int m0 = blockIdx.y * 64, n0 = blockIdx.x * 128;
  const int tid = threadIdx.x, lane = tid & 31, wv = tid >> 5;
  const int wm = wv & 1, wn = wv >> 1;            // 2x4 wave grid; 32x32 per wave

  v8f c00 = {0.f,0.f,0.f,0.f,0.f,0.f,0.f,0.f};
  v8f c01 = {0.f,0.f,0.f,0.f,0.f,0.f,0.f,0.f};
  v8f c10 = {0.f,0.f,0.f,0.f,0.f,0.f,0.f,0.f};
  v8f c11 = {0.f,0.f,0.f,0.f,0.f,0.f,0.f,0.f};

  const int ra = tid >> 2, ka = (tid & 3) * 8;    // A: 64 rows x 32 k, 8 f/thr
  const int rb = tid >> 1, kb = (tid & 1) * 16;   // B NT: 128 rows x 32 k, 16 f/thr

  for (int k0 = 0; k0 < K; k0 += 32) {
    {
      const float* s = A + (long long)(m0 + ra) * lda + (k0 + ka);
      bf16* dp = &As[ra * LSTR + ka];
      #pragma unroll
      for (int j = 0; j < 8; ++j) dp[j] = (bf16)s[j];
      if (k0 + 32 < K) __builtin_prefetch(s + 32, 0, 0);
    }
    if (TRANSB) {
      const float* s = Bm + (long long)(n0 + rb) * ldb + (k0 + kb);
      bf16* dp = &Bs[rb * LSTR + kb];
      #pragma unroll
      for (int j = 0; j < 16; ++j) dp[j] = (bf16)s[j];
      if (k0 + 32 < K) __builtin_prefetch(s + 32, 0, 0);
    } else {
      const int kk = tid >> 3, nn = (tid & 7) * 16; // 32 k-rows x 128 n, transpose
      const float* s = Bm + (long long)(k0 + kk) * ldb + (n0 + nn);
      #pragma unroll
      for (int j = 0; j < 16; ++j) Bs[(nn + j) * LSTR + kk] = (bf16)s[j];
    }
    __syncthreads();

    const int r16 = lane & 15, hlf = lane >> 4;
    v16bf a0 = fragA(&As[(wm * 32 +      r16) * LSTR], hlf);
    v16bf a1 = fragA(&As[(wm * 32 + 16 + r16) * LSTR], hlf);
    v16bf b0 = fragB(&Bs[(wn * 32 +      r16) * LSTR], hlf);
    v16bf b1 = fragB(&Bs[(wn * 32 + 16 + r16) * LSTR], hlf);

    c00 = __builtin_amdgcn_wmma_f32_16x16x32_bf16(false, a0, false, b0, (short)0, c00, false, false);
    c01 = __builtin_amdgcn_wmma_f32_16x16x32_bf16(false, a0, false, b1, (short)0, c01, false, false);
    c10 = __builtin_amdgcn_wmma_f32_16x16x32_bf16(false, a1, false, b0, (short)0, c10, false, false);
    c11 = __builtin_amdgcn_wmma_f32_16x16x32_bf16(false, a1, false, b1, (short)0, c11, false, false);
    __syncthreads();
  }

  // C layout: VGPR r -> row r (lanes 0-15) / r+8 (lanes 16-31); col = lane&15
  const int r16 = lane & 15, hlf = lane >> 4;
  const long long col0 = n0 + wn * 32 + r16;
  const long long col1 = col0 + 16;
  const long long row0 = m0 + wm * 32 + hlf * 8;
  const long long row1 = row0 + 16;
  const float bi0 = bias ? bias[col0] : 0.f;
  const float bi1 = bias ? bias[col1] : 0.f;
  #pragma unroll
  for (int r = 0; r < 8; ++r) {
    C[(row0 + r) * ldc + col0] = c00[r] * alpha + bi0;
    C[(row0 + r) * ldc + col1] = c01[r] * alpha + bi1;
    C[(row1 + r) * ldc + col0] = c10[r] * alpha + bi0;
    C[(row1 + r) * ldc + col1] = c11[r] * alpha + bi1;
  }
}

// ---------------- row kernels ----------------
__global__ __launch_bounds__(TPB)
void krms(const float* __restrict__ x, const float* __restrict__ w,
          float* __restrict__ y, int Dl)
{
  __shared__ float sd[TPB];
  long long row = blockIdx.x;
  const float* xr = x + row * Dl;
  float s = 0.f;
  for (int i = threadIdx.x; i < Dl; i += TPB) { float v = xr[i]; s += v * v; }
  s = bsum(s, sd);
  float sc = rsqrtf(s / (float)Dl + EPSf);
  for (int i = threadIdx.x; i < Dl; i += TPB) y[row * Dl + i] = xr[i] * sc * w[i];
}

__global__ __launch_bounds__(TPB)
void ksoftmax(float* __restrict__ p, int N)
{
  __shared__ float sd[TPB];
  long long row = blockIdx.x;
  float* pr = p + row * (long long)N;
  float m = -1e30f;
  for (int i = threadIdx.x; i < N; i += TPB) m = fmaxf(m, pr[i]);
  m = bmaxr(m, sd);
  float s = 0.f;
  for (int i = threadIdx.x; i < N; i += TPB) { float e = __expf(pr[i] - m); pr[i] = e; s += e; }
  s = bsum(s, sd);
  float inv = 1.0f / s;
  for (int i = threadIdx.x; i < N; i += TPB) pr[i] *= inv;
}

// ---------------- elementwise / small kernels ----------------
__global__ void kscale_kv(const float* __restrict__ rh, const float* __restrict__ conf,
                          float* __restrict__ akv)
{
  long long idx = (long long)blockIdx.x * TPB + threadIdx.x;  // over NR*B*SKV*D
  int d = (int)(idx % Dm); long long r = idx / Dm;
  int s = (int)(r % SKVs); r /= SKVs;
  int b = (int)(r % BBt);  int n = (int)(r / BBt);
  akv[(((long long)b * KVT) + (long long)n * SKVs + s) * Dm + d] =
      rh[idx] * conf[b * NRs + n];
}

__global__ void kattn_mean(const float* __restrict__ sc, float* __restrict__ out)
{
  long long idx = (long long)blockIdx.x * TPB + threadIdx.x;  // over B*SQ*KVT
  int n = (int)(idx % KVT); long long r = idx / KVT;
  int q = (int)(r % SQs);  int b = (int)(r / SQs);
  float s = 0.f;
  #pragma unroll
  for (int h = 0; h < Hh; ++h)
    s += sc[(((long long)(b * Hh + h) * SQs) + q) * KVT + n];
  out[idx] = s * (1.0f / Hh);
}

__global__ void krope(float* __restrict__ t)   // [B*SQ, D], heads of 128
{
  long long idx = (long long)blockIdx.x * TPB + threadIdx.x;  // rows*H*64
  int j = (int)(idx & 63);
  long long rh = idx >> 6;
  int h = (int)(rh & (Hh - 1));
  long long row = rh >> 3;
  int s = (int)(row & (SQs - 1));
  float* p = t + row * Dm + h * HDd;
  float inv = __powf(10000.0f, -(float)j * (1.0f / 64.0f));
  float f = (float)s * inv;
  float sn, cs; __sincosf(f, &sn, &cs);
  float a = p[j], b = p[j + 64];
  p[j]      = a * cs - b * sn;
  p[j + 64] = b * cs + a * sn;
}

__global__ void kadd(float* __restrict__ x, const float* __restrict__ y)
{ long long i = (long long)blockIdx.x * TPB + threadIdx.x; x[i] += y[i]; }

__global__ void kadd_bcast(float* __restrict__ x, const float* __restrict__ a)
{ long long i = (long long)blockIdx.x * TPB + threadIdx.x;
  int d = (int)(i % Dm); int b = (int)(i / ((long long)SQs * Dm));
  x[i] += a[b * Dm + d]; }

__global__ void ksilumul(float* __restrict__ g, const float* __restrict__ u)
{ long long i = (long long)blockIdx.x * TPB + threadIdx.x;
  float x = g[i]; g[i] = (x / (1.0f + __expf(-x))) * u[i]; }

__global__ void kgelu(float* __restrict__ x, int n)
{ long long i = (long long)blockIdx.x * TPB + threadIdx.x;
  if (i < n) { float v = x[i]; x[i] = 0.5f * v * (1.0f + erff(v * 0.70710678118f)); } }

__global__ void ksumm(const float* __restrict__ rh, float* __restrict__ sm)
{ // sm[b, n*D+d] = mean_s rh[n,b,s,d]
  long long idx = (long long)blockIdx.x * TPB + threadIdx.x;  // B*NR*D
  int d = (int)(idx % Dm); long long r = idx / Dm;
  int n = (int)(r % NRs);  int b = (int)(r / NRs);
  const float* p = rh + (((long long)n * BBt + b) * SKVs) * Dm + d;
  float s = 0.f;
  for (int k = 0; k < SKVs; ++k) s += p[(long long)k * Dm];
  sm[((long long)b * NRs + n) * Dm + d] = s * (1.0f / SKVs);
}

__global__ void kpooled(const float* __restrict__ x, float* __restrict__ out)
{ long long idx = (long long)blockIdx.x * TPB + threadIdx.x;  // B*D
  int d = (int)(idx % Dm); int b = (int)(idx / Dm);
  const float* p = x + ((long long)b * SQs) * Dm + d;
  float s = 0.f;
  for (int k = 0; k < SQs; ++k) s += p[(long long)k * Dm];
  out[idx] = s * (1.0f / SQs);
}

__global__ __launch_bounds__(TPB)
void kgemv(const float* __restrict__ A, int lda, const float* __restrict__ W, int ldw,
           const float* __restrict__ bias, float* __restrict__ C, int ldc, int K)
{
  __shared__ float sd[TPB];
  int n = blockIdx.x, m = blockIdx.y;
  const float* a = A + (long long)m * lda;
  const float* w = W + (long long)n * ldw;
  float s = 0.f;
  for (int k = threadIdx.x; k < K; k += TPB) s += a[k] * w[k];
  s = bsum(s, sd);
  if (threadIdx.x == 0) C[(long long)m * ldc + n] = s + (bias ? bias[n] : 0.f);
}

// ---------------- host-side GEMM launcher ----------------
static inline void gemm(hipStream_t st, bool tb,
  const float* A, long long lda, long long sAb, long long sAh,
  const float* Bm, long long ldb, long long sBb, long long sBh,
  float* C, long long ldc, long long sCb, long long sCh,
  int M, int N, int K, int batch, int Hb, const float* bias, float alpha)
{
  dim3 g(N / 128, M / 64, batch), blk(TPB);
  if (tb) kgemm<true ><<<g, blk, 0, st>>>(A,lda,sAb,sAh, Bm,ldb,sBb,sBh, C,ldc,sCb,sCh, K,bias,alpha,Hb);
  else    kgemm<false><<<g, blk, 0, st>>>(A,lda,sAb,sAh, Bm,ldb,sBb,sBh, C,ldc,sCb,sCh, K,bias,alpha,Hb);
}

} // namespace

// ===========================================================================
extern "C" void kernel_launch(void* const* d_in, const int* in_sizes, int n_in,
                              void* d_out, int out_size, void* d_ws, size_t ws_size,
                              hipStream_t stream)
{
  const float* rh      = (const float*)d_in[0];
  const float* conf    = (const float*)d_in[1];
  const float* qh      = (const float*)d_in[3];
  const float* ca_in_w = (const float*)d_in[4];
  const float* ca_in_b = (const float*)d_in[5];
  const float* ca_out_w= (const float*)d_in[6];
  const float* ca_out_b= (const float*)d_in[7];
  const float* cross_nw= (const float*)d_in[8];
  const float* ag_w1   = (const float*)d_in[9];
  const float* ag_b1   = (const float*)d_in[10];
  const float* ag_w2   = (const float*)d_in[11];
  const float* ag_b2   = (const float*)d_in[12];
  const float* ag_nw   = (const float*)d_in[13];
  const float* gen_n1  = (const float*)d_in[14];
  const float* gwq     = (const float*)d_in[15];
  const float* gwk     = (const float*)d_in[16];
  const float* gwv     = (const float*)d_in[17];
  const float* gwo     = (const float*)d_in[18];
  const float* gen_n2  = (const float*)d_in[19];
  const float* gwg     = (const float*)d_in[20];
  const float* gwu     = (const float*)d_in[21];
  const float* gwd     = (const float*)d_in[22];
  const float* gen_nw  = (const float*)d_in[23];
  const float* lm      = (const float*)d_in[24];
  const float* m_w1    = (const float*)d_in[25];
  const float* m_b1    = (const float*)d_in[26];
  const float* m_w2    = (const float*)d_in[27];
  const float* m_b2    = (const float*)d_in[28];

  // output layout: logits | meta | attn_weights
  float* logits = (float*)d_out;
  float* meta_o = logits + (long long)BBt * SQs * VOC;          // 65,536,000
  float* attn_o = meta_o + BBt * 3;

  // workspace layout (floats)
  float* w = (float*)d_ws;
  const long long SZX = (long long)BBt * SQs * Dm;              // 2,097,152
  float* X    = w;                 w += SZX;
  float* XN   = w;                 w += SZX;
  float* Q    = w;                 w += SZX;
  float* Kb   = w;                 w += 2 * SZX;                // up to 4096 rows
  float* Vb   = w;                 w += 2 * SZX;
  float* AT   = w;                 w += SZX;                    // attn out (pre-proj)
  float* PR   = w;                 w += SZX;                    // proj out
  float* AKV  = w;                 w += 2 * SZX;                // weighted kv
  float* SC   = w;                 w += (long long)BBt * Hh * SQs * KVT; // 33,554,432
  float* GATE = w;                 w += (long long)BBt * SQs * FFd;
  float* UP   = w;                 w += (long long)BBt * SQs * FFd;
  float* SUMM = w;                 w += BBt * NRs * Dm;
  float* AGH  = w;                 w += BBt * 2 * Dm;
  float* AGR  = w;                 w += BBt * Dm;
  float* POOL = w;                 w += BBt * Dm;
  float* MH   = w;                 w += BBt * (Dm / 4);
  (void)ws_size; (void)in_sizes; (void)n_in; (void)out_size;

  const int ROWS = BBt * SQs;   // 2048

  // 1) confidence-weighted KV
  kscale_kv<<<(NRs * BBt * SKVs * Dm) / TPB, TPB, 0, stream>>>(rh, conf, AKV);
  // 2) x = query_hidden
  hipMemcpyAsync(X, qh, sizeof(float) * SZX, hipMemcpyDeviceToDevice, stream);

  // 3) cross-attention layers
  for (int i = 0; i < 2; ++i) {
    krms<<<ROWS, TPB, 0, stream>>>(X, cross_nw + i * Dm, XN, Dm);
    const float* inw = ca_in_w + (long long)i * 3 * Dm * Dm;
    const float* inb = ca_in_b + (long long)i * 3 * Dm;
    gemm(stream, true, XN, Dm,0,0, inw,                    Dm,0,0, Q,  Dm,0,0, ROWS,    Dm, Dm, 1,1, inb,        1.f);
    gemm(stream, true, AKV,Dm,0,0, inw + (long long)Dm*Dm, Dm,0,0, Kb, Dm,0,0, BBt*KVT, Dm, Dm, 1,1, inb + Dm,   1.f);
    gemm(stream, true, AKV,Dm,0,0, inw + 2ll*Dm*Dm,        Dm,0,0, Vb, Dm,0,0, BBt*KVT, Dm, Dm, 1,1, inb + 2*Dm, 1.f);
    // scores[b,h] = q @ k^T / sqrt(hd)
    gemm(stream, true,  Q,  Dm, (long long)SQs*Dm, HDd,
                        Kb, Dm, (long long)KVT*Dm, HDd,
                        SC, KVT, (long long)Hh*SQs*KVT, (long long)SQs*KVT,
                        SQs, KVT, HDd, BBt*Hh, Hh, nullptr, ATT_SCALE);
    ksoftmax<<<BBt * Hh * SQs, TPB, 0, stream>>>(SC, KVT);
    if (i == 1)
      kattn_mean<<<(BBt * SQs * KVT) / TPB, TPB, 0, stream>>>(SC, attn_o);
    // o[b,h] = p @ v  (NN)
    gemm(stream, false, SC, KVT, (long long)Hh*SQs*KVT, (long long)SQs*KVT,
                        Vb, Dm,  (long long)KVT*Dm, HDd,
                        AT, Dm,  (long long)SQs*Dm, HDd,
                        SQs, HDd, KVT, BBt*Hh, Hh, nullptr, 1.f);
    gemm(stream, true, AT, Dm,0,0, ca_out_w + (long long)i*Dm*Dm, Dm,0,0,
         PR, Dm,0,0, ROWS, Dm, Dm, 1,1, ca_out_b + i*Dm, 1.f);
    kadd<<<(ROWS * Dm) / TPB, TPB, 0, stream>>>(X, PR);
  }

  // 4) agreement MLP
  ksumm<<<(BBt * NRs * Dm) / TPB, TPB, 0, stream>>>(rh, SUMM);
  { dim3 g(2 * Dm, BBt); kgemv<<<g, TPB, 0, stream>>>(SUMM, NRs*Dm, ag_w1, NRs*Dm, ag_b1, AGH, 2*Dm, NRs*Dm); }
  kgelu<<<(BBt * 2 * Dm) / TPB, TPB, 0, stream>>>(AGH, BBt * 2 * Dm);
  { dim3 g(Dm, BBt); kgemv<<<g, TPB, 0, stream>>>(AGH, 2*Dm, ag_w2, 2*Dm, ag_b2, AGR, Dm, 2*Dm); }
  krms<<<BBt, TPB, 0, stream>>>(AGR, ag_nw, AGR, Dm);
  kadd_bcast<<<(ROWS * Dm) / TPB, TPB, 0, stream>>>(X, AGR);

  // 5) generator layers (RoPE self-attn + SwiGLU FFN)
  for (int l = 0; l < NLl; ++l) {
    krms<<<ROWS, TPB, 0, stream>>>(X, gen_n1 + l * Dm, XN, Dm);
    const long long wo_off = (long long)l * Dm * Dm;
    gemm(stream, true, XN, Dm,0,0, gwq + wo_off, Dm,0,0, Q,  Dm,0,0, ROWS, Dm, Dm, 1,1, nullptr, 1.f);
    gemm(stream, true, XN, Dm,0,0, gwk + wo_off, Dm,0,0, Kb, Dm,0,0, ROWS, Dm, Dm, 1,1, nullptr, 1.f);
    gemm(stream, true, XN, Dm,0,0, gwv + wo_off, Dm,0,0, Vb, Dm,0,0, ROWS, Dm, Dm, 1,1, nullptr, 1.f);
    krope<<<(ROWS * Hh * 64) / TPB, TPB, 0, stream>>>(Q);
    krope<<<(ROWS * Hh * 64) / TPB, TPB, 0, stream>>>(Kb);
    gemm(stream, true,  Q,  Dm, (long long)SQs*Dm, HDd,
                        Kb, Dm, (long long)SQs*Dm, HDd,
                        SC, SQs, (long long)Hh*SQs*SQs, (long long)SQs*SQs,
                        SQs, SQs, HDd, BBt*Hh, Hh, nullptr, ATT_SCALE);
    ksoftmax<<<BBt * Hh * SQs, TPB, 0, stream>>>(SC, SQs);
    gemm(stream, false, SC, SQs, (long long)Hh*SQs*SQs, (long long)SQs*SQs,
                        Vb, Dm,  (long long)SQs*Dm, HDd,
                        AT, Dm,  (long long)SQs*Dm, HDd,
                        SQs, HDd, SQs, BBt*Hh, Hh, nullptr, 1.f);
    gemm(stream, true, AT, Dm,0,0, gwo + wo_off, Dm,0,0, PR, Dm,0,0, ROWS, Dm, Dm, 1,1, nullptr, 1.f);
    kadd<<<(ROWS * Dm) / TPB, TPB, 0, stream>>>(X, PR);

    krms<<<ROWS, TPB, 0, stream>>>(X, gen_n2 + l * Dm, XN, Dm);
    const long long wf_off = (long long)l * FFd * Dm;
    gemm(stream, true, XN, Dm,0,0, gwg + wf_off, Dm,0,0, GATE, FFd,0,0, ROWS, FFd, Dm, 1,1, nullptr, 1.f);
    gemm(stream, true, XN, Dm,0,0, gwu + wf_off, Dm,0,0, UP,   FFd,0,0, ROWS, FFd, Dm, 1,1, nullptr, 1.f);
    ksilumul<<<(ROWS * FFd) / TPB, TPB, 0, stream>>>(GATE, UP);
    gemm(stream, true, GATE, FFd,0,0, gwd + (long long)l*Dm*FFd, FFd,0,0,
         PR, Dm,0,0, ROWS, Dm, FFd, 1,1, nullptr, 1.f);
    kadd<<<(ROWS * Dm) / TPB, TPB, 0, stream>>>(X, PR);
  }

  // 6) final norm, LM head, meta head
  krms<<<ROWS, TPB, 0, stream>>>(X, gen_nw, XN, Dm);
  gemm(stream, true, XN, Dm,0,0, lm, Dm,0,0, logits, VOC,0,0, ROWS, VOC, Dm, 1,1, nullptr, 1.f);
  kpooled<<<(BBt * Dm) / TPB, TPB, 0, stream>>>(XN, POOL);
  { dim3 g(Dm / 4, BBt); kgemv<<<g, TPB, 0, stream>>>(POOL, Dm, m_w1, Dm, m_b1, MH, Dm/4, Dm); }
  kgelu<<<(BBt * (Dm/4) + TPB - 1) / TPB, TPB, 0, stream>>>(MH, BBt * (Dm/4));
  { dim3 g(3, BBt); kgemv<<<g, TPB, 0, stream>>>(MH, Dm/4, m_w2, Dm/4, m_b2, meta_o, 3, Dm/4); }
}
